// Attention_1159641170177
// MI455X (gfx1250) — matrix operand order repaired and checked by reading, not compiled, not run on hardware
//
#include <hip/hip_runtime.h>

typedef unsigned short u16;
typedef __attribute__((ext_vector_type(16))) __bf16 v16bf;
typedef __attribute__((ext_vector_type(8)))  float  v8f;
typedef __attribute__((ext_vector_type(4)))  unsigned int u32x4;
typedef __attribute__((ext_vector_type(4)))  int i32x4;

union Frag { v16bf bf; u32x4 u[2]; };

__device__ __forceinline__ u16 f2bf(float f) {
  unsigned int u = __builtin_bit_cast(unsigned int, f);
  u += 0x7fffu + ((u >> 16) & 1u);          // round-to-nearest-even
  return (u16)(u >> 16);
}

// ---------------------------------------------------------------------------
// CDNA5 async global->LDS staging (ASYNCcnt path), with sync fallback.
// Builtin signature (from hipcc diagnostic): param0 = int4 addrspace(1)*,
// i.e. (global_src, lds_dst, imm offset, imm cpol).
// ---------------------------------------------------------------------------
#if __has_builtin(__builtin_amdgcn_global_load_async_to_lds_b128)
#define HAVE_ASYNC 1
#else
#define HAVE_ASYNC 0
#endif

__device__ __forceinline__ void stage16(const u16* g, u16* l) {
#if HAVE_ASYNC
  __builtin_amdgcn_global_load_async_to_lds_b128(
      (__attribute__((address_space(1))) i32x4*)g,
      (__attribute__((address_space(3))) i32x4*)l, 0, 0);
#else
  *(u32x4*)l = *(const u32x4*)g;
#endif
}

__device__ __forceinline__ void stage_wait() {
#if HAVE_ASYNC
#if __has_builtin(__builtin_amdgcn_s_wait_asynccnt)
  __builtin_amdgcn_s_wait_asynccnt(0);
#else
  asm volatile("s_wait_asynccnt 0x0" ::: "memory");
#endif
#endif
}

// ---------------------------------------------------------------------------
// 16-lane xor butterflies via v_permlane16_b32 (VALU) instead of ds_bpermute.
// ---------------------------------------------------------------------------
#if __has_builtin(__builtin_amdgcn_permlane16)
#define HAVE_PERMLANE 1
#else
#define HAVE_PERMLANE 0
#endif

template <int M, unsigned LO, unsigned HI>
__device__ __forceinline__ float bfly(float v) {
#if HAVE_PERMLANE
  unsigned u = __builtin_bit_cast(unsigned, v);
  u = __builtin_amdgcn_permlane16(u, u, LO, HI, false, false);
  return __builtin_bit_cast(float, u);
#else
  return __shfl_xor(v, M, 32);
#endif
}

__device__ __forceinline__ float red_max16(float v) {
  v = fmaxf(v, bfly<1, 0x67452301u, 0xEFCDAB89u>(v));
  v = fmaxf(v, bfly<2, 0x54761032u, 0xDCFE98BAu>(v));
  v = fmaxf(v, bfly<4, 0x32107654u, 0xBA98FEDCu>(v));
  v = fmaxf(v, bfly<8, 0xFEDCBA98u, 0x76543210u>(v));
  return v;
}
__device__ __forceinline__ float red_sum16(float v) {
  v += bfly<1, 0x67452301u, 0xEFCDAB89u>(v);
  v += bfly<2, 0x54761032u, 0xDCFE98BAu>(v);
  v += bfly<4, 0x32107654u, 0xBA98FEDCu>(v);
  v += bfly<8, 0xFEDCBA98u, 0x76543210u>(v);
  return v;
}

// ---------------------------------------------------------------------------
// f32 -> bf16 elementwise convert
// ---------------------------------------------------------------------------
__global__ __launch_bounds__(256) void cvt_bf16(const float* __restrict__ in,
                                                u16* __restrict__ out, int n) {
  int stride = gridDim.x * blockDim.x;
  for (int i = blockIdx.x * blockDim.x + threadIdx.x; i < n; i += stride)
    out[i] = f2bf(in[i]);
}

// ---------------------------------------------------------------------------
// f32 [K,N] -> bf16 [N,K] tiled transpose (so GEMMs run "NT")
// ---------------------------------------------------------------------------
__global__ __launch_bounds__(256) void transpose_cvt(const float* __restrict__ in,
                                                     u16* __restrict__ out,
                                                     int K, int N) {
  __shared__ u16 tile[32][33];
  const int kb = blockIdx.y * 32, nb = blockIdx.x * 32;
  const int tx = threadIdx.x & 31, ty = threadIdx.x >> 5;  // 32 x 8
#pragma unroll
  for (int i = 0; i < 32; i += 8)
    tile[ty + i][tx] = f2bf(in[(size_t)(kb + ty + i) * N + nb + tx]);
  __syncthreads();
#pragma unroll
  for (int i = 0; i < 32; i += 8)
    out[(size_t)(nb + ty + i) * K + kb + tx] = tile[tx][ty + i];
}

// ---------------------------------------------------------------------------
// Tiled WMMA NT-GEMM: D = A[M,K] * Bt[N,K]^T + bias
// Block tile 128x256, BK=64, 8 waves (2x4), wave tile 64x64 (4x4 wmma tiles).
// Double-buffered LDS, async global->LDS staging.
// MODE 0: qkv epilogue (bias, q-scale, scatter to q[B,H,N,D], k[B,H,N,D],
//         vt[B,H,D,N] as bf16).   MODE 1: f32 out + bias.
// ---------------------------------------------------------------------------
__device__ __forceinline__ void gemm_stage(const u16* __restrict__ A,
                                           const u16* __restrict__ Bt,
                                           u16* As, u16* Bs,
                                           int m0, int n0, int k0, int Ksz, int t) {
#pragma unroll
  for (int i = 0; i < 4; i++) {                  // A: 128 rows x 64
    int c = t + i * 256, row = c >> 3, seg = c & 7;
    stage16(A + (size_t)(m0 + row) * Ksz + k0 + seg * 8, As + row * 64 + seg * 8);
  }
#pragma unroll
  for (int i = 0; i < 8; i++) {                  // Bt: 256 rows x 64
    int c = t + i * 256, row = c >> 3, seg = c & 7;
    stage16(Bt + (size_t)(n0 + row) * Ksz + k0 + seg * 8, Bs + row * 64 + seg * 8);
  }
}

template <int MODE>
__global__ __launch_bounds__(256) void gemm_nt(
    const u16* __restrict__ A, const u16* __restrict__ Bt,
    const float* __restrict__ bias, float* __restrict__ outf,
    u16* __restrict__ qo, u16* __restrict__ ko, u16* __restrict__ vto,
    int Nsz, int Ksz) {
  __shared__ u16 As[2][128 * 64];
  __shared__ u16 Bs[2][256 * 64];
  const int t = threadIdx.x, w = t >> 5, lane = t & 31;
  const int half = lane >> 4, ln = lane & 15;
  const int wm = w >> 2, wn = w & 3;                      // 2 x 4 wave grid
  const int m0 = blockIdx.y * 128, n0 = blockIdx.x * 256;

  v8f acc[4][4] = {};

  gemm_stage(A, Bt, As[0], Bs[0], m0, n0, 0, Ksz, t);
  stage_wait();
  __syncthreads();

  int buf = 0;
  for (int k0 = 0; k0 < Ksz; k0 += 64) {
    if (k0 + 64 < Ksz)
      gemm_stage(A, Bt, As[buf ^ 1], Bs[buf ^ 1], m0, n0, k0 + 64, Ksz, t);

    const u16* Ab = As[buf];
    const u16* Bb = Bs[buf];
#pragma unroll
    for (int kk = 0; kk < 2; kk++) {
      Frag a[4];
#pragma unroll
      for (int mt = 0; mt < 4; mt++) {                    // A frags: M=lane%16
        const u16* p = Ab + (wm * 64 + mt * 16 + ln) * 64 + kk * 32 + 8 * half;
        a[mt].u[0] = *(const u32x4*)p;
        a[mt].u[1] = *(const u32x4*)(p + 16);
      }
#pragma unroll
      for (int nt = 0; nt < 4; nt++) {                    // B frags: N=lane%16
        Frag bb;
        const u16* p = Bb + (wn * 64 + nt * 16 + ln) * 64 + kk * 32 + 8 * half;
        bb.u[0] = *(const u32x4*)p;
        bb.u[1] = *(const u32x4*)(p + 16);
#pragma unroll
        for (int mt = 0; mt < 4; mt++)
          acc[mt][nt] = __builtin_amdgcn_wmma_f32_16x16x32_bf16(
              false, a[mt].bf, false, bb.bf, (short)0, acc[mt][nt], false, false);
      }
    }
    stage_wait();
    __syncthreads();
    buf ^= 1;
  }

#pragma unroll
  for (int mt = 0; mt < 4; mt++)
#pragma unroll
    for (int nt = 0; nt < 4; nt++) {
      const int n = n0 + wn * 64 + nt * 16 + ln;
      const float bv = bias[n];
#pragma unroll
      for (int r = 0; r < 8; r++) {
        const int m = m0 + wm * 64 + mt * 16 + r + 8 * half;
        float v = acc[mt][nt][r] + bv;
        if (MODE == 0) {
          const int three = n >> 10, rem = n & 1023, hh = rem >> 6, d = rem & 63;
          const int bb_ = m >> 10, tok = m & 1023;
          const size_t nd = ((size_t)(bb_ * 16 + hh)) * 1024 + tok;
          if (three == 0)
            qo[nd * 64 + d] = f2bf(v * 0.125f);           // q * D^-0.5
          else if (three == 1)
            ko[nd * 64 + d] = f2bf(v);
          else
            vto[(((size_t)(bb_ * 16 + hh)) * 64 + d) * 1024 + tok] = f2bf(v);
        } else {
          outf[(size_t)m * Nsz + n] = v;
        }
      }
    }
}

// ---------------------------------------------------------------------------
// Flash attention: block = (b, h, 128-row q tile); 8 waves x 16 q rows.
// KV tiles of 128, double-buffered async staging, online softmax, WMMA bf16.
// ---------------------------------------------------------------------------
__device__ __forceinline__ void attn_stage(const u16* __restrict__ kg,
                                           const u16* __restrict__ vg,
                                           u16* ksb, u16* vtsb, int kv, int t) {
#pragma unroll
  for (int i = 0; i < 4; i++) {                  // K: 128 rows x 64 (n', d)
    int c = t + i * 256, row = c >> 3, seg = c & 7;
    stage16(kg + (size_t)(kv + row) * 64 + seg * 8, ksb + row * 64 + seg * 8);
  }
#pragma unroll
  for (int i = 0; i < 4; i++) {                  // V^T: 64 rows x 128 (d, n')
    int c = t + i * 256, row = c >> 4, seg = c & 15;
    stage16(vg + (size_t)row * 1024 + kv + seg * 8, vtsb + row * 128 + seg * 8);
  }
}

__global__ __launch_bounds__(256) void attn_kernel(
    const u16* __restrict__ q, const u16* __restrict__ k,
    const u16* __restrict__ vt, u16* __restrict__ attnout) {
  __shared__ u16 qs[128 * 64];
  __shared__ u16 ks[2][128 * 64];
  __shared__ u16 vts[2][64 * 128];
  __shared__ u16 ps[8 * 16 * 128];   // per-wave P scratch (C-layout -> A-layout)

  const int t = threadIdx.x, w = t >> 5, lane = t & 31;
  const int half = lane >> 4, ln = lane & 15;
  const int b = blockIdx.z, h = blockIdx.y, q0 = blockIdx.x * 128;
  const size_t bh = (size_t)(b * 16 + h);
  const u16* qg = q + bh * 1024 * 64;
  const u16* kg = k + bh * 1024 * 64;
  const u16* vg = vt + bh * 64 * 1024;

#pragma unroll
  for (int i = 0; i < 4; i++) {                          // stage full q tile
    int c = t + i * 256, row = c >> 3, seg = c & 7;
    stage16(qg + (size_t)(q0 + row) * 64 + seg * 8, qs + row * 64 + seg * 8);
  }
  attn_stage(kg, vg, ks[0], vts[0], 0, t);
  stage_wait();
  __syncthreads();

  float mrun[8], lrun[8];
  v8f o[4] = {};
#pragma unroll
  for (int r = 0; r < 8; r++) { mrun[r] = -3.0e38f; lrun[r] = 0.f; }

  int buf = 0;
  for (int kv = 0; kv < 1024; kv += 128) {
    if (kv + 128 < 1024)
      attn_stage(kg, vg, ks[buf ^ 1], vts[buf ^ 1], kv + 128, t);

    // S = Q * K^T   (16 x 128 per wave)
    v8f s[8] = {};
#pragma unroll
    for (int kk = 0; kk < 2; kk++) {
      Frag a;
      const u16* ap = qs + (w * 16 + ln) * 64 + kk * 32 + 8 * half;
      a.u[0] = *(const u32x4*)ap;
      a.u[1] = *(const u32x4*)(ap + 16);
#pragma unroll
      for (int j = 0; j < 8; j++) {
        Frag bb;
        const u16* bp = ks[buf] + (j * 16 + ln) * 64 + kk * 32 + 8 * half;
        bb.u[0] = *(const u32x4*)bp;
        bb.u[1] = *(const u32x4*)(bp + 16);
        s[j] = __builtin_amdgcn_wmma_f32_16x16x32_bf16(
            false, a.bf, false, bb.bf, (short)0, s[j], false, false);
      }
    }

    // online softmax: row = r + 8*half lives across the 16 lanes of the half
    float alpha[8];
#pragma unroll
    for (int r = 0; r < 8; r++) {
      float v = fmaxf(fmaxf(fmaxf(s[0][r], s[1][r]), fmaxf(s[2][r], s[3][r])),
                      fmaxf(fmaxf(s[4][r], s[5][r]), fmaxf(s[6][r], s[7][r])));
      v = red_max16(v);
      float mnew = fmaxf(mrun[r], v);
      alpha[r] = __expf(mrun[r] - mnew);
      mrun[r] = mnew;
    }
#pragma unroll
    for (int r = 0; r < 8; r++) {
      float sum = 0.f;
#pragma unroll
      for (int j = 0; j < 8; j++) {
        float p = __expf(s[j][r] - mrun[r]);
        s[j][r] = p;
        sum += p;
      }
      sum = red_sum16(sum);
      lrun[r] = lrun[r] * alpha[r] + sum;
    }
#pragma unroll
    for (int j = 0; j < 4; j++)
#pragma unroll
      for (int r = 0; r < 8; r++) o[j][r] *= alpha[r];

    // redistribute P (C-layout) -> A-fragment layout via per-wave LDS
    u16* pw = ps + w * 2048;
#pragma unroll
    for (int j = 0; j < 8; j++)
#pragma unroll
      for (int r = 0; r < 8; r++)
        pw[(r + 8 * half) * 128 + j * 16 + ln] = f2bf(s[j][r]);
    __syncthreads();

    // O += P * V
#pragma unroll
    for (int kk = 0; kk < 4; kk++) {
      Frag a;
      const u16* ap = ps + w * 2048 + ln * 128 + kk * 32 + 8 * half;
      a.u[0] = *(const u32x4*)ap;
      a.u[1] = *(const u32x4*)(ap + 16);
#pragma unroll
      for (int j = 0; j < 4; j++) {
        Frag bb;
        const u16* bp = vts[buf] + (j * 16 + ln) * 128 + kk * 32 + 8 * half;
        bb.u[0] = *(const u32x4*)bp;
        bb.u[1] = *(const u32x4*)(bp + 16);
        o[j] = __builtin_amdgcn_wmma_f32_16x16x32_bf16(
            false, a.bf, false, bb.bf, (short)0, o[j], false, false);
      }
    }
    stage_wait();
    __syncthreads();
    buf ^= 1;
  }

  float inv[8];
#pragma unroll
  for (int r = 0; r < 8; r++) inv[r] = 1.0f / lrun[r];
#pragma unroll
  for (int j = 0; j < 4; j++)
#pragma unroll
    for (int r = 0; r < 8; r++) {
      const int row = q0 + w * 16 + r + 8 * half;
      attnout[((size_t)b * 1024 + row) * 1024 + h * 64 + j * 16 + ln] =
          f2bf(o[j][r] * inv[r]);
    }
}

// ---------------------------------------------------------------------------
extern "C" void kernel_launch(void* const* d_in, const int* in_sizes, int n_in,
                              void* d_out, int out_size, void* d_ws, size_t ws_size,
                              hipStream_t stream) {
  (void)in_sizes; (void)n_in; (void)out_size; (void)ws_size;
  const float* x     = (const float*)d_in[0];  // [4,1024,1024]
  const float* wqkv  = (const float*)d_in[1];  // [1024,3072]
  const float* bqkv  = (const float*)d_in[2];  // [3072]
  const float* wproj = (const float*)d_in[3];  // [1024,1024]
  const float* bproj = (const float*)d_in[4];  // [1024]
  float* out = (float*)d_out;

  char* ws = (char*)d_ws;                       // 48 MB total
  u16* xbf    = (u16*)(ws);                     //  8 MB  x bf16 [4096,1024]
  u16* wqkvt  = (u16*)(ws + (8u  << 20));       //  6 MB  wqkv^T bf16 [3072,1024]
  u16* wprojt = (u16*)(ws + (14u << 20));       //  2 MB  wproj^T bf16 [1024,1024]
  u16* qb     = (u16*)(ws + (16u << 20));       //  8 MB  q  [B,H,N,D]
  u16* kb     = (u16*)(ws + (24u << 20));       //  8 MB  k  [B,H,N,D]
  u16* vtb    = (u16*)(ws + (32u << 20));       //  8 MB  v^T[B,H,D,N]
  u16* attnb  = (u16*)(ws + (40u << 20));       //  8 MB  attn out bf16 [4096,1024]

  cvt_bf16<<<2048, 256, 0, stream>>>(x, xbf, 4096 * 1024);
  transpose_cvt<<<dim3(96, 32), 256, 0, stream>>>(wqkv, wqkvt, 1024, 3072);
  transpose_cvt<<<dim3(32, 32), 256, 0, stream>>>(wproj, wprojt, 1024, 1024);

  // qkv = x @ wqkv + b, scatter into q (scaled), k, v^T
  gemm_nt<0><<<dim3(12, 32), 256, 0, stream>>>(xbf, wqkvt, bqkv, nullptr,
                                               qb, kb, vtb, 3072, 1024);
  // attention
  attn_kernel<<<dim3(8, 16, 4), 256, 0, stream>>>(qb, kb, vtb, attnb);
  // out = attn @ wproj + b
  gemm_nt<1><<<dim3(4, 32), 256, 0, stream>>>(attnb, wprojt, bproj, out,
                                              nullptr, nullptr, nullptr, 1024, 1024);
}
